// MultiheadAttention_1065151889808
// MI455X (gfx1250) — matrix operand order repaired and checked by reading, not compiled, not run on hardware
//
#include <hip/hip_runtime.h>

#define E_DIM 1024
#define T_SEQ 2048
#define NH 16
#define HD 64
#define BSZ 2

typedef __attribute__((ext_vector_type(16))) __bf16        v16bf;
typedef __attribute__((ext_vector_type(8)))  float         v8f;
typedef __attribute__((ext_vector_type(4)))  unsigned int  u32x4;
typedef __attribute__((ext_vector_type(4)))  float         f32x4;

union Frag { v16bf v; unsigned short u[16]; u32x4 q4[2]; };

__device__ __forceinline__ unsigned short f32_to_bf16(float f) {
  unsigned int u = __builtin_bit_cast(unsigned int, f);
  u += 0x7fffu + ((u >> 16) & 1u);           // round-to-nearest-even
  return (unsigned short)(u >> 16);
}
__device__ __forceinline__ float f16b_to_f32(unsigned short b) {
  return (float)__builtin_bit_cast(_Float16, b);
}
__device__ __forceinline__ unsigned short f32_to_f16b(float f) {
  return __builtin_bit_cast(unsigned short, (_Float16)f);
}

__device__ __forceinline__ v8f wmma_bf16(const Frag& a, const Frag& b, v8f c) {
  return __builtin_amdgcn_wmma_f32_16x16x32_bf16(
      /*neg_a=*/false, a.v, /*neg_b=*/false, b.v,
      /*c_mod=*/(short)0, c, /*reuse_a=*/false, /*reuse_b=*/false);
}

// A fragment: 16x32 bf16 tile at (rows 0..15 of base, cols k0..k0+31), row stride ld.
// ISA layout: lanes 0-15 = rows, elems 0-7 -> K=half*8+j, elems 8-15 -> K=16+half*8+j.
__device__ __forceinline__ Frag load_a_bf16(const unsigned short* base, int ld, int lane, int k0) {
  int r = lane & 15, half = lane >> 4;
  const unsigned short* p = base + (size_t)r * ld + k0 + half * 8;
  Frag f;
  f.q4[0] = *(const u32x4*)(p);        // K = half*8 + 0..7
  f.q4[1] = *(const u32x4*)(p + 16);   // K = 16 + half*8 + 0..7
  return f;
}

// B fragment: B[k][n] = M[n0+n][k0+k] with M row-major bf16; base points at M[n0][k0].
// ISA layout: lane n (0-15), kh = lane>>4, elems 0-7 -> K = kh*8+j, elems 8-15 -> K = 16+kh*8+j.
__device__ __forceinline__ Frag load_bT_bf16(const unsigned short* base, int ld, int lane) {
  int n = lane & 15, kh = lane >> 4;
  const unsigned short* p = base + (size_t)n * ld + kh * 8;
  Frag f;
  f.q4[0] = *(const u32x4*)(p);
  f.q4[1] = *(const u32x4*)(p + 16);
  return f;
}

// ---------------- Kernel 0: one-shot f32 -> bf16 conversion ----------------
// Pure memory-bound pass; removes all conversion VALU from the GEMM hot loops.
__global__ void cvt_bf16_kernel(const float* __restrict__ src,
                                unsigned short* __restrict__ dst, int n4) {
  int i = blockIdx.x * blockDim.x + threadIdx.x;
  if (i < n4) {
    f32x4 v = *(const f32x4*)(src + (size_t)i * 4);
    unsigned short o0 = f32_to_bf16(v[0]), o1 = f32_to_bf16(v[1]);
    unsigned short o2 = f32_to_bf16(v[2]), o3 = f32_to_bf16(v[3]);
    unsigned int lo = (unsigned int)o0 | ((unsigned int)o1 << 16);
    unsigned int hi = (unsigned int)o2 | ((unsigned int)o3 << 16);
    *(uint2*)(dst + (size_t)i * 4) = make_uint2(lo, hi);
  }
}

// ---------------- Kernel 1: fused QKV projection (pure bf16) ----------------
// y = x @ W^T + b; q scaled by 1/sqrt(HD). q,k stored bf16 [B*T, E]; v stored
// transposed bf16 [B, H, D, T] so P@V B-operands are contiguous loads.
__global__ void qkv_proj_kernel(const unsigned short* __restrict__ xb,
                                const unsigned short* __restrict__ Wq,
                                const unsigned short* __restrict__ Wk,
                                const unsigned short* __restrict__ Wv,
                                const float* __restrict__ bq,
                                const float* __restrict__ bk,
                                const float* __restrict__ bv,
                                unsigned short* __restrict__ q_ws,
                                unsigned short* __restrict__ k_ws,
                                unsigned short* __restrict__ vt_ws) {
  int tid = threadIdx.x, wave = tid >> 5, lane = tid & 31;
  int m0 = (blockIdx.x >> 3) * 16;                 // 256 row tiles over B*T
  int n0 = ((blockIdx.x & 7) * 8 + wave) * 16;     // 64 col tiles over E
  v8f accq = {}, acck = {}, accv = {};
  const unsigned short* arow = xb + (size_t)m0 * E_DIM;
  const unsigned short* wqp = Wq + (size_t)n0 * E_DIM;
  const unsigned short* wkp = Wk + (size_t)n0 * E_DIM;
  const unsigned short* wvp = Wv + (size_t)n0 * E_DIM;
  for (int kc = 0; kc < E_DIM; kc += 32) {
    Frag A  = load_a_bf16(arow, E_DIM, lane, kc);  // shared across 3 GEMMs
    Frag Bq = load_bT_bf16(wqp + kc, E_DIM, lane);
    accq = wmma_bf16(A, Bq, accq);
    Frag Bk = load_bT_bf16(wkp + kc, E_DIM, lane);
    acck = wmma_bf16(A, Bk, acck);
    Frag Bv = load_bT_bf16(wvp + kc, E_DIM, lane);
    accv = wmma_bf16(A, Bv, accv);
  }
  int n = lane & 15;
  float bqv = bq[n0 + n], bkv = bk[n0 + n], bvv = bv[n0 + n];
#pragma unroll
  for (int r = 0; r < 8; ++r) {
    int M = ((lane >> 4) << 3) + r;
    int gm = m0 + M;                    // row in [0, B*T)
    int gn = n0 + n;                    // col in [0, E)
    q_ws[(size_t)gm * E_DIM + gn] = f32_to_bf16((accq[r] + bqv) * 0.125f);
    k_ws[(size_t)gm * E_DIM + gn] = f32_to_bf16(acck[r] + bkv);
    int bb = gm >> 11, tt = gm & (T_SEQ - 1);
    int hh = gn >> 6,  dd = gn & (HD - 1);
    vt_ws[(((size_t)bb * NH + hh) * HD + dd) * T_SEQ + tt] = f32_to_bf16(accv[r] + bvv);
  }
}

// ---------------- Kernel 2: fused attention + head-averaged probs ----------------
// One block per (b, 16-row tile); loops over all heads so the head-average
// accumulates in LDS and hits HBM exactly once (no atomics).
__global__ void attn_kernel(const unsigned short* __restrict__ q_ws,
                            const unsigned short* __restrict__ k_ws,
                            const unsigned short* __restrict__ vt_ws,
                            const float* __restrict__ bias,
                            unsigned short* __restrict__ attn_ws,
                            float* __restrict__ avg_out) {
  extern __shared__ char smem[];
  float*          avg_acc = (float*)smem;                      // 16*2048 f32 = 128 KB
  unsigned short* s_u16   = (unsigned short*)(smem + 131072);  // 16*2048 f16/bf16 = 64 KB
  unsigned short* q_lds   = (unsigned short*)(smem + 196608);  // 16*64 bf16 = 2 KB
  float*          o_red   = (float*)(smem + 198656);           // 2*4*16*16 f32 = 8 KB
  float*          red     = (float*)(smem + 206848);           // 16*16 f32
  float*          m_row   = (float*)(smem + 207872);           // 16 f32
  float*          l_row   = (float*)(smem + 207936);           // 16 f32

  int tid = threadIdx.x, wave = tid >> 5, lane = tid & 31;
  int b  = blockIdx.x >> 7;               // 128 row tiles per batch
  int m0 = (blockIdx.x & 127) * 16;

  for (int i = tid; i < 16 * T_SEQ; i += 256) avg_acc[i] = 0.f;

  const unsigned short* qp = q_ws  + (size_t)b * T_SEQ * E_DIM;
  const unsigned short* kp = k_ws  + (size_t)b * T_SEQ * E_DIM;
  const unsigned short* vp = vt_ws + (size_t)b * NH * HD * T_SEQ;

  int rrow = tid >> 4, rsub = tid & 15;   // softmax reduction: 16 threads per row
  int nt = wave & 3,  kh = wave >> 2;     // P@V: 4 N-tiles x 2 K-halves

  for (int h = 0; h < NH; ++h) {
    __syncthreads();
    // stage Q tile [16 x 64] for this head
    for (int i = tid; i < 16 * HD; i += 256) {
      int r = i >> 6, c = i & 63;
      q_lds[i] = qp[(size_t)(m0 + r) * E_DIM + h * HD + c];
    }
    __syncthreads();

    // S = q @ k^T + bias, stored as f16 in LDS (each wave: 16 column tiles)
    const float* brow = bias + (((size_t)b * NH + h) * T_SEQ + m0) * T_SEQ;
    for (int ct = wave; ct < T_SEQ / 16; ct += 8) {
      int col0 = ct * 16;
      v8f acc = {};
#pragma unroll
      for (int kc = 0; kc < HD; kc += 32) {
        Frag A = load_a_bf16(q_lds, HD, lane, kc);
        Frag B = load_bT_bf16(kp + (size_t)col0 * E_DIM + h * HD + kc, E_DIM, lane);
        acc = wmma_bf16(A, B, acc);
      }
      int nn = lane & 15;
#pragma unroll
      for (int r = 0; r < 8; ++r) {
        int M = ((lane >> 4) << 3) + r;
        float sv = acc[r] + brow[(size_t)M * T_SEQ + col0 + nn];
        s_u16[M * T_SEQ + col0 + nn] = f32_to_f16b(sv);
      }
    }
    __syncthreads();

    // rowwise softmax stats in fp32
    const unsigned short* srow = s_u16 + rrow * T_SEQ;
    int c0 = rsub * 128;
    float mloc = -3.0e38f;
    for (int c = c0; c < c0 + 128; ++c) mloc = fmaxf(mloc, f16b_to_f32(srow[c]));
    red[rrow * 16 + rsub] = mloc;
    __syncthreads();
    if (rsub == 0) {
      float m = red[rrow * 16];
      for (int i = 1; i < 16; ++i) m = fmaxf(m, red[rrow * 16 + i]);
      m_row[rrow] = m;
    }
    __syncthreads();
    float mrow = m_row[rrow];
    float sloc = 0.f;
    for (int c = c0; c < c0 + 128; ++c) sloc += __expf(f16b_to_f32(srow[c]) - mrow);
    red[rrow * 16 + rsub] = sloc;
    __syncthreads();
    if (rsub == 0) {
      float s = 0.f;
      for (int i = 0; i < 16; ++i) s += red[rrow * 16 + i];
      l_row[rrow] = s;
    }
    __syncthreads();
    float rl = 1.0f / l_row[rrow];
    unsigned short* swrow = s_u16 + rrow * T_SEQ;
    float* arow2 = avg_acc + rrow * T_SEQ;
    for (int c = c0; c < c0 + 128; ++c) {    // in-place f16 -> prob(bf16), same slot
      float p = __expf(f16b_to_f32(swrow[c]) - mrow) * rl;
      arow2[c] += p;
      swrow[c] = f32_to_bf16(p);
    }
    __syncthreads();

    // O = P @ V  (wave: N-tile nt, K-half kh; partials reduced via LDS)
    v8f oacc = {};
    for (int kc = kh * (T_SEQ / 2); kc < (kh + 1) * (T_SEQ / 2); kc += 32) {
      Frag A = load_a_bf16(s_u16, T_SEQ, lane, kc);
      Frag B = load_bT_bf16(vp + ((size_t)h * HD + nt * 16) * T_SEQ + kc, T_SEQ, lane);
      oacc = wmma_bf16(A, B, oacc);
    }
    float* op = o_red + (kh * 4 + nt) * 256;
    int nn = lane & 15;
#pragma unroll
    for (int r = 0; r < 8; ++r) {
      int M = ((lane >> 4) << 3) + r;
      op[M * 16 + nn] = oacc[r];
    }
    __syncthreads();
    if (kh == 0) {
#pragma unroll
      for (int r = 0; r < 8; ++r) {
        int M = ((lane >> 4) << 3) + r;
        float v = o_red[nt * 256 + M * 16 + nn] + o_red[(4 + nt) * 256 + M * 16 + nn];
        attn_ws[((size_t)b * T_SEQ + m0 + M) * E_DIM + h * HD + nt * 16 + nn] =
            f32_to_bf16(v);
      }
    }
  }
  __syncthreads();
  // head-averaged attention weights -> second output
  for (int i = tid; i < 16 * T_SEQ; i += 256) {
    int r = i >> 11, c = i & (T_SEQ - 1);
    avg_out[((size_t)b * T_SEQ + m0 + r) * T_SEQ + c] = avg_acc[i] * (1.0f / NH);
  }
}

// ---------------- Kernel 3: output projection (pure bf16) ----------------
__global__ void out_proj_kernel(const unsigned short* __restrict__ attn_ws,
                                const unsigned short* __restrict__ Wo,
                                const float* __restrict__ bo,
                                float* __restrict__ out) {
  int tid = threadIdx.x, wave = tid >> 5, lane = tid & 31;
  int m0 = (blockIdx.x >> 3) * 16;
  int n0 = ((blockIdx.x & 7) * 8 + wave) * 16;
  v8f acc = {};
  const unsigned short* arow = attn_ws + (size_t)m0 * E_DIM;
  const unsigned short* wop = Wo + (size_t)n0 * E_DIM;
  for (int kc = 0; kc < E_DIM; kc += 32) {
    Frag A = load_a_bf16(arow, E_DIM, lane, kc);
    Frag B = load_bT_bf16(wop + kc, E_DIM, lane);
    acc = wmma_bf16(A, B, acc);
  }
  int n = lane & 15;
  float bov = bo[n0 + n];
#pragma unroll
  for (int r = 0; r < 8; ++r) {
    int M = ((lane >> 4) << 3) + r;
    out[(size_t)(m0 + M) * E_DIM + n0 + n] = acc[r] + bov;
  }
}

extern "C" void kernel_launch(void* const* d_in, const int* in_sizes, int n_in,
                              void* d_out, int out_size, void* d_ws, size_t ws_size,
                              hipStream_t stream) {
  const float* x    = (const float*)d_in[0];   // query (k/v args ignored: self-attn)
  const float* bias = (const float*)d_in[3];
  const float* Wq   = (const float*)d_in[4];
  const float* bq   = (const float*)d_in[5];
  const float* Wk   = (const float*)d_in[6];
  const float* bk   = (const float*)d_in[7];
  const float* Wv   = (const float*)d_in[8];
  const float* bv   = (const float*)d_in[9];
  const float* Wo   = (const float*)d_in[10];
  const float* bo   = (const float*)d_in[11];

  const size_t NTOK = (size_t)BSZ * T_SEQ;     // 4096
  const size_t ACT  = NTOK * E_DIM;            // 4 Mi elements
  const size_t WSZ  = (size_t)E_DIM * E_DIM;   // 1 Mi elements

  unsigned short* q_ws    = (unsigned short*)d_ws;
  unsigned short* k_ws    = q_ws  + ACT;
  unsigned short* vt_ws   = k_ws  + ACT;
  unsigned short* xb      = vt_ws + ACT;       // x in bf16; dead after kernel 1
  unsigned short* attn_ws = xb;                // ...so attn output re-uses it
  unsigned short* wq_b    = xb + ACT;
  unsigned short* wk_b    = wq_b + WSZ;
  unsigned short* wv_b    = wk_b + WSZ;
  unsigned short* wo_b    = wv_b + WSZ;

  float* out = (float*)d_out;
  float* avg = out + ACT;                      // second output, [B, T, T]

  // one-shot f32 -> bf16 conversions (memory-bound)
  cvt_bf16_kernel<<<(int)(ACT / 4 + 255) / 256, 256, 0, stream>>>(x, xb, (int)(ACT / 4));
  cvt_bf16_kernel<<<(int)(WSZ / 4 + 255) / 256, 256, 0, stream>>>(Wq, wq_b, (int)(WSZ / 4));
  cvt_bf16_kernel<<<(int)(WSZ / 4 + 255) / 256, 256, 0, stream>>>(Wk, wk_b, (int)(WSZ / 4));
  cvt_bf16_kernel<<<(int)(WSZ / 4 + 255) / 256, 256, 0, stream>>>(Wv, wv_b, (int)(WSZ / 4));
  cvt_bf16_kernel<<<(int)(WSZ / 4 + 255) / 256, 256, 0, stream>>>(Wo, wo_b, (int)(WSZ / 4));

  qkv_proj_kernel<<<2048, 256, 0, stream>>>(xb, wq_b, wk_b, wv_b, bq, bk, bv,
                                            q_ws, k_ws, vt_ws);
  size_t smem2 = 208000;                       // ~203 KB of the 320 KB WGP LDS
  attn_kernel<<<BSZ * (T_SEQ / 16), 256, smem2, stream>>>(q_ws, k_ws, vt_ws, bias,
                                                          attn_ws, avg);
  out_proj_kernel<<<2048, 256, 0, stream>>>(attn_ws, wo_b, bo, out);
}
